// SparseMatrixEquivariantNetwork_54666343743773
// MI455X (gfx1250) — matrix operand order repaired and checked by reading, not compiled, not run on hardware
//
#include <hip/hip_runtime.h>
#include <math.h>

// ---------------------------------------------------------------------------
// CDNA5 (gfx1250, wave32) implementation of the sparse equivariant network.
//
// Precision plan (bandwidth-bound: ~1 GB of edge traffic/pass @ 23.3 TB/s):
//   * edge activations A stored as f16 (halves HBM traffic, native WMMA operand)
//   * all accumulation, node tensors, pooling, stats in fp32
//   * per-edge GEMMs via v_wmma_f32_16x16x32_f16 (probe-confirmed builtin)
// ---------------------------------------------------------------------------

typedef _Float16 v16h __attribute__((ext_vector_type(16)));
typedef float    v8f  __attribute__((ext_vector_type(8)));

union Frag {            // one WMMA 16-half operand fragment per lane (32 bytes)
    uint4     u[2];
    v16h      v;
    _Float16  h[16];
};

union H8 { uint4 u; _Float16 h[8]; };

// ---------------------------------------------------------------------------
// generic helpers
// ---------------------------------------------------------------------------
__global__ __launch_bounds__(256) void fill0(float* __restrict__ p, long n) {
    long t = (long)blockIdx.x * blockDim.x + threadIdx.x;
    if (t < n) p[t] = 0.f;
}

// Pack fp32 weight [CIN, COUT] into WMMA B-fragment order (f16).
// B 32x16 layout (derived from ISA sparse-B doc): lane n = lane&15,
// K = kt*32 + (r>>2)*16 + (lane>>4)*8 + 2*(r&3) (+1), VGPR r in 0..7. Zero-pad K >= CIN.
__global__ __launch_bounds__(256) void pack_wmma_b(const float* __restrict__ W,
                                                   _Float16* __restrict__ out,
                                                   int CIN, int COUT) {
    int ntiles = COUT / 16;
    int ktiles = (CIN + 31) / 32;
    int total  = ktiles * ntiles * 256;         // 32 lanes * 8 dwords per frag
    int t = blockIdx.x * blockDim.x + threadIdx.x;
    if (t >= total) return;
    int r    = t & 7;
    int lane = (t >> 3) & 31;
    int f    = t >> 8;
    int nt = f % ntiles, kt = f / ntiles;
    int n  = nt * 16 + (lane & 15);
    int hi = lane >> 4;
    int k0 = kt * 32 + (r >> 2) * 16 + hi * 8 + 2 * (r & 3);
    _Float16 h0 = (k0     < CIN) ? (_Float16)W[(long)k0 * COUT + n]       : (_Float16)0.f;
    _Float16 h1 = (k0 + 1 < CIN) ? (_Float16)W[(long)(k0 + 1) * COUT + n] : (_Float16)0.f;
    out[(long)t * 2]     = h0;
    out[(long)t * 2 + 1] = h1;
}

// ---------------------------------------------------------------------------
// pooling (segment sums + global pools)
// ---------------------------------------------------------------------------
__global__ __launch_bounds__(256) void pool_edges_f32c7(const float* __restrict__ A,
                                                        const int* __restrict__ row,
                                                        const int* __restrict__ col,
                                                        float* __restrict__ rs,
                                                        float* __restrict__ cs,
                                                        float* __restrict__ ta, int E) {
    __shared__ float sta[7];
    for (int i = threadIdx.x; i < 7; i += blockDim.x) sta[i] = 0.f;
    __syncthreads();
    int e = blockIdx.x * blockDim.x + threadIdx.x;
    if (e < E) {
        int rn = row[e], cn = col[e];
        const float* ap = A + (long)e * 7;
        #pragma unroll
        for (int c = 0; c < 7; ++c) {
            float v = ap[c];
            atomicAdd(&rs[(long)rn * 7 + c], v);
            atomicAdd(&cs[(long)cn * 7 + c], v);
            atomicAdd(&sta[c], v);
        }
    }
    __syncthreads();
    for (int i = threadIdx.x; i < 7; i += blockDim.x) atomicAdd(&ta[i], sta[i]);
}

template <int CIN>
__global__ __launch_bounds__(256) void pool_edges_f16(const _Float16* __restrict__ A,
                                                      const int* __restrict__ row,
                                                      const int* __restrict__ col,
                                                      float* __restrict__ rs,
                                                      float* __restrict__ cs,
                                                      float* __restrict__ ta, int E) {
    __shared__ float sta[CIN];
    for (int i = threadIdx.x; i < CIN; i += blockDim.x) sta[i] = 0.f;
    __syncthreads();
    int e = blockIdx.x * blockDim.x + threadIdx.x;
    if (e < E) {
        int rn = row[e], cn = col[e];
        const uint4* ap = (const uint4*)(A + (long)e * CIN);
        #pragma unroll
        for (int g = 0; g < CIN / 8; ++g) {
            H8 x; x.u = ap[g];
            #pragma unroll
            for (int j = 0; j < 8; ++j) {
                float v = (float)x.h[j];
                int c = g * 8 + j;
                atomicAdd(&rs[(long)rn * CIN + c], v);
                atomicAdd(&cs[(long)cn * CIN + c], v);
                atomicAdd(&sta[c], v);
            }
        }
    }
    __syncthreads();
    for (int i = threadIdx.x; i < CIN; i += blockDim.x) atomicAdd(&ta[i], sta[i]);
}

__global__ __launch_bounds__(256) void pool_nodes(const float* __restrict__ D,
                                                  float* __restrict__ td, int N, int CIN) {
    __shared__ float s[64];
    for (int i = threadIdx.x; i < CIN; i += blockDim.x) s[i] = 0.f;
    __syncthreads();
    int n = blockIdx.x * blockDim.x + threadIdx.x;
    if (n < N) {
        const float* dp = D + (long)n * CIN;
        for (int c = 0; c < CIN; ++c) atomicAdd(&s[c], dp[c]);
    }
    __syncthreads();
    for (int i = threadIdx.x; i < CIN; i += blockDim.x) atomicAdd(&td[i], s[i]);
}

// out[n] = sum_ci ta[ci]*Wa[ci][n] + td[ci]*Wb[ci][n] + bias[n]
__global__ void cvec_kernel(const float* __restrict__ ta, const float* __restrict__ td,
                            const float* __restrict__ Wa, const float* __restrict__ Wb,
                            const float* __restrict__ bias, float* __restrict__ out,
                            int CIN, int COUT) {
    int n = threadIdx.x;
    if (n >= COUT) return;
    float a = bias[n];
    for (int ci = 0; ci < CIN; ++ci)
        a += ta[ci] * Wa[(long)ci * COUT + n] + td[ci] * Wb[(long)ci * COUT + n];
    out[n] = a;
}

// ---------------------------------------------------------------------------
// per-node broadcast terms + raw D_out
// ---------------------------------------------------------------------------
__global__ __launch_bounds__(256) void node_terms_kernel(
        const float* __restrict__ rs, const float* __restrict__ cs, const float* __restrict__ D,
        const float* __restrict__ Waa2, const float* __restrict__ Waa4, const float* __restrict__ Wda0,
        const float* __restrict__ Waa3, const float* __restrict__ Waa5, const float* __restrict__ Wda1,
        const float* __restrict__ Wad0, const float* __restrict__ Wad1, const float* __restrict__ Wdd0,
        const float* __restrict__ cvecD,
        float* __restrict__ Rrow, float* __restrict__ Rcol, float* __restrict__ Draw,
        int N, int CIN, int COUT) {
    long t = (long)blockIdx.x * blockDim.x + threadIdx.x;
    if (t >= (long)N * COUT) return;
    int n = (int)(t / COUT), j = (int)(t % COUT);
    const float* rsp = rs + (long)n * CIN;
    const float* csp = cs + (long)n * CIN;
    const float* dp  = D  + (long)n * CIN;
    float ar = 0.f, ac = 0.f, ad = 0.f;
    for (int ci = 0; ci < CIN; ++ci) {
        float r = rsp[ci], c = csp[ci], d = dp[ci];
        long w = (long)ci * COUT + j;
        ar += r * Waa2[w] + c * Waa4[w] + d * Wda0[w];
        ac += r * Waa3[w] + c * Waa5[w] + d * Wda1[w];
        ad += r * Wad0[w] + c * Wad1[w] + d * Wdd0[w];
    }
    Rrow[t] = ar;
    Rcol[t] = ac;
    Draw[t] = ad + cvecD[j];
}

__global__ __launch_bounds__(256) void relu_stats_nodes(float* __restrict__ D,
                                                        float* __restrict__ dstat,
                                                        long total, int co) {
    __shared__ float s[64], q[64];
    for (int i = threadIdx.x; i < co; i += blockDim.x) { s[i] = 0.f; q[i] = 0.f; }
    __syncthreads();
    long t = (long)blockIdx.x * blockDim.x + threadIdx.x;
    if (t < total) {
        float v = fmaxf(D[t], 0.f);
        D[t] = v;
        int c = (int)(t % co);
        atomicAdd(&s[c], v);
        atomicAdd(&q[c], v * v);
    }
    __syncthreads();
    for (int i = threadIdx.x; i < co; i += blockDim.x) {
        atomicAdd(&dstat[i],      s[i]);
        atomicAdd(&dstat[64 + i], q[i]);
    }
}

// ---------------------------------------------------------------------------
// edge WMMA kernel: Araw = relu(A@W0 + A[tidx]@W1 + Rrow[row] + Rcol[col] + cvecA)
// one wave32 per 16-edge tile; accumulates per-channel sum/sumsq for the norm.
// ---------------------------------------------------------------------------
template <int CIN, int COUT, bool F16IN>
__global__ __launch_bounds__(128) void edge_wmma_kernel(
        const void* __restrict__ Aprev,
        const _Float16* __restrict__ Wp0, const _Float16* __restrict__ Wp1,
        const float* __restrict__ Rrow, const float* __restrict__ Rcol,
        const float* __restrict__ cvecA,
        const int* __restrict__ erow, const int* __restrict__ ecol,
        const int* __restrict__ tidx,
        _Float16* __restrict__ Araw, float* __restrict__ astat) {
    constexpr int KT = (CIN + 31) / 32;
    constexpr int NT = COUT / 16;

    __shared__ float ssum[COUT], ssq[COUT];
    for (int i = threadIdx.x; i < COUT; i += blockDim.x) { ssum[i] = 0.f; ssq[i] = 0.f; }
    __syncthreads();

    const int wave = threadIdx.x >> 5;
    const int lane = threadIdx.x & 31;
    const int tile = blockIdx.x * 4 + wave;
    const long e0  = (long)tile * 16;
    const int  m   = lane & 15;
    const int  hi  = lane >> 4;
    const int  kb  = hi * 8;

    __builtin_prefetch(Wp0, 0, 3);   // global_prefetch_b8 of packed weights
    __builtin_prefetch(Wp1, 0, 3);

    const long ep = (long)tidx[e0 + m];   // transpose partner row for lane's edge

    Frag fa[KT], fp[KT];
    if constexpr (F16IN) {
        const _Float16* rA = (const _Float16*)Aprev + (e0 + m) * (long)CIN;
        const _Float16* rP = (const _Float16*)Aprev + ep * (long)CIN;
        #pragma unroll
        for (int kt = 0; kt < KT; ++kt) {
            fa[kt].u[0] = *(const uint4*)(rA + kt * 32 + kb);
            fa[kt].u[1] = *(const uint4*)(rA + kt * 32 + 16 + kb);
            fp[kt].u[0] = *(const uint4*)(rP + kt * 32 + kb);
            fp[kt].u[1] = *(const uint4*)(rP + kt * 32 + 16 + kb);
        }
    } else {
        // layer 1: CIN==7 fp32 input, zero-padded to K=32
        #pragma unroll
        for (int kt = 0; kt < KT; ++kt) {
            fa[kt].u[0] = make_uint4(0, 0, 0, 0); fa[kt].u[1] = make_uint4(0, 0, 0, 0);
            fp[kt].u[0] = make_uint4(0, 0, 0, 0); fp[kt].u[1] = make_uint4(0, 0, 0, 0);
        }
        if (hi == 0) {      // lanes 0-15 hold K=0..7 in VGPRs 0..3 -> h[0..7]
            const float* rA = (const float*)Aprev + (e0 + m) * 7;
            const float* rP = (const float*)Aprev + ep * 7;
            #pragma unroll
            for (int c = 0; c < 7; ++c) {
                fa[0].h[c] = (_Float16)rA[c];
                fp[0].h[c] = (_Float16)rP[c];
            }
        }
    }

    const Frag* B0 = (const Frag*)Wp0;
    const Frag* B1 = (const Frag*)Wp1;
    const v8f zerov = {0.f, 0.f, 0.f, 0.f, 0.f, 0.f, 0.f, 0.f};
    v8f acc[NT];
    #pragma unroll
    for (int nt = 0; nt < NT; ++nt) acc[nt] = zerov;

    #pragma unroll
    for (int kt = 0; kt < KT; ++kt) {
        #pragma unroll
        for (int nt = 0; nt < NT; ++nt) {
            Frag b0 = B0[(kt * NT + nt) * 32 + lane];
            Frag b1 = B1[(kt * NT + nt) * 32 + lane];
            acc[nt] = __builtin_amdgcn_wmma_f32_16x16x32_f16(
                false, fa[kt].v, false, b0.v, (short)0, acc[nt], false, false);
            acc[nt] = __builtin_amdgcn_wmma_f32_16x16x32_f16(
                false, fp[kt].v, false, b1.v, (short)0, acc[nt], false, false);
        }
    }

    // epilogue: broadcast terms, bias, relu, f16 store, per-channel stats
    #pragma unroll
    for (int nt = 0; nt < NT; ++nt) {
        const int n = nt * 16 + m;
        const float cv = cvecA[n];
        float s = 0.f, q = 0.f;
        #pragma unroll
        for (int r = 0; r < 8; ++r) {
            const long e = e0 + r + 8 * hi;
            float v = acc[nt][r]
                    + Rrow[(long)erow[e] * COUT + n]
                    + Rcol[(long)ecol[e] * COUT + n] + cv;
            v = fmaxf(v, 0.f);
            Araw[e * COUT + n] = (_Float16)v;
            s += v; q += v * v;
        }
        atomicAdd(&ssum[n], s);   // ds_add_f32
        atomicAdd(&ssq[n],  q);
    }
    __syncthreads();
    for (int i = threadIdx.x; i < COUT; i += blockDim.x) {
        atomicAdd(&astat[i],      ssum[i]);
        atomicAdd(&astat[64 + i], ssq[i]);
    }
}

// ---------------------------------------------------------------------------
// normalization
// ---------------------------------------------------------------------------
__global__ void finalize_stats(const float* __restrict__ astat, const float* __restrict__ dstat,
                               float* __restrict__ statfin, float cntA, float cntD, int co) {
    int c = threadIdx.x;
    if (c >= co) return;
    float mA = astat[c] / cntA;
    float vA = astat[64 + c] / cntA - mA * mA;
    statfin[c]      = mA;
    statfin[64 + c] = rsqrtf(vA + 1e-5f);
    float mD = dstat[c] / cntD;
    float vD = dstat[64 + c] / cntD - mD * mD;
    statfin[128 + c] = mD;
    statfin[192 + c] = rsqrtf(vD + 1e-5f);
}

__global__ __launch_bounds__(256) void norm_edges(_Float16* __restrict__ A,
                                                  const float* __restrict__ statfin,
                                                  long total8, int co) {
    long t = (long)blockIdx.x * blockDim.x + threadIdx.x;
    if (t >= total8) return;
    H8 x; x.u = ((const uint4*)A)[t];
    int c0 = (int)((t * 8) % co);
    #pragma unroll
    for (int j = 0; j < 8; ++j) {
        float v = (float)x.h[j];
        v = (v - statfin[c0 + j]) * statfin[64 + c0 + j];
        x.h[j] = (_Float16)v;
    }
    ((uint4*)A)[t] = x.u;
}

__global__ __launch_bounds__(256) void norm_nodes(float* __restrict__ D,
                                                  const float* __restrict__ statfin,
                                                  long total, int co) {
    long t = (long)blockIdx.x * blockDim.x + threadIdx.x;
    if (t >= total) return;
    int c = (int)(t % co);
    D[t] = (D[t] - statfin[128 + c]) * statfin[192 + c];
}

// ---------------------------------------------------------------------------
// final head: logits + softmax (Cin fixed at 32, 7 classes)
// ---------------------------------------------------------------------------
__global__ __launch_bounds__(128) void logits_kernel(
        const float* __restrict__ rs, const float* __restrict__ cs, const float* __restrict__ D,
        const float* __restrict__ W0, const float* __restrict__ W1, const float* __restrict__ W2,
        const float* __restrict__ cvec, float* __restrict__ out, int N) {
    int n = blockIdx.x * blockDim.x + threadIdx.x;
    if (n >= N) return;
    float acc[7];
    #pragma unroll
    for (int j = 0; j < 7; ++j) acc[j] = cvec[j];
    const float* rsp = rs + (long)n * 32;
    const float* csp = cs + (long)n * 32;
    const float* dp  = D  + (long)n * 32;
    for (int ci = 0; ci < 32; ++ci) {
        float r = rsp[ci], c = csp[ci], d = dp[ci];
        #pragma unroll
        for (int j = 0; j < 7; ++j)
            acc[j] += r * W0[ci * 7 + j] + c * W1[ci * 7 + j] + d * W2[ci * 7 + j];
    }
    float mx = acc[0];
    #pragma unroll
    for (int j = 1; j < 7; ++j) mx = fmaxf(mx, acc[j]);
    float s = 0.f;
    #pragma unroll
    for (int j = 0; j < 7; ++j) { acc[j] = __expf(acc[j] - mx); s += acc[j]; }
    float inv = 1.f / s;
    #pragma unroll
    for (int j = 0; j < 7; ++j) out[(long)n * 7 + j] = acc[j] * inv;
}

// ---------------------------------------------------------------------------
// host orchestration
// ---------------------------------------------------------------------------
static inline unsigned cdiv(long a, long b) { return (unsigned)((a + b - 1) / b); }

extern "C" void kernel_launch(void* const* d_in, const int* in_sizes, int n_in,
                              void* d_out, int out_size, void* d_ws, size_t ws_size,
                              hipStream_t stream) {
    (void)n_in; (void)out_size; (void)ws_size;

    const float* x_adj  = (const float*)d_in[0];
    const float* x_diag = (const float*)d_in[1];
    const int*   erow   = (const int*)d_in[2];
    const int*   ecol   = (const int*)d_in[3];
    const int*   tidx   = (const int*)d_in[4];
    const int E = in_sizes[2];
    const int N = in_sizes[1] / 7;

    const float* Waa[3] = {(const float*)d_in[5],  (const float*)d_in[11], (const float*)d_in[17]};
    const float* Wad[3] = {(const float*)d_in[6],  (const float*)d_in[12], (const float*)d_in[18]};
    const float* Wda[3] = {(const float*)d_in[7],  (const float*)d_in[13], (const float*)d_in[19]};
    const float* Wdd[3] = {(const float*)d_in[8],  (const float*)d_in[14], (const float*)d_in[20]};
    const float* ba[3]  = {(const float*)d_in[9],  (const float*)d_in[15], (const float*)d_in[21]};
    const float* bd[3]  = {(const float*)d_in[10], (const float*)d_in[16], (const float*)d_in[22]};
    const float* W4ad = (const float*)d_in[23];
    const float* W4dd = (const float*)d_in[24];
    const float* b4d  = (const float*)d_in[25];
    const int ci_[3] = {7, 32, 64}, co_[3] = {32, 64, 32};

    // ---- carve workspace ----
    char* wp = (char*)d_ws;
    auto carve = [&](size_t b) { char* r = wp; wp += (b + 255) & ~(size_t)255; return r; };
    _Float16* Abuf[2];
    Abuf[0] = (_Float16*)carve((size_t)E * 64 * sizeof(_Float16));
    Abuf[1] = (_Float16*)carve((size_t)E * 64 * sizeof(_Float16));
    const long zn = (long)2 * N * 64 + 384;               // rs|cs|ta|td|astat|dstat
    float* Z    = (float*)carve((size_t)zn * 4);
    float* rs   = Z;
    float* cs   = Z + (size_t)N * 64;
    float* ta   = cs + (size_t)N * 64;
    float* td   = ta + 64;
    float* astat = td + 64;
    float* dstat = astat + 128;
    float* Rrow = (float*)carve((size_t)N * 64 * 4);
    float* Rcol = (float*)carve((size_t)N * 64 * 4);
    float* Dbuf[2];
    Dbuf[0] = (float*)carve((size_t)N * 64 * 4);
    Dbuf[1] = (float*)carve((size_t)N * 64 * 4);
    float* statfin = (float*)carve(256 * 4);
    float* cvA = (float*)carve(64 * 4);
    float* cvD = (float*)carve(64 * 4);
    _Float16* packW[6];
    for (int i = 0; i < 6; ++i) packW[i] = (_Float16*)carve(2048 * sizeof(_Float16));

    // ---- pre-pack edge GEMM weights into WMMA B-fragment order ----
    for (int l = 0; l < 3; ++l) {
        int cin = ci_[l], cout = co_[l];
        int tot = ((cin + 31) / 32) * (cout / 16) * 256;
        pack_wmma_b<<<cdiv(tot, 256), 256, 0, stream>>>(Waa[l], packW[2 * l], cin, cout);
        pack_wmma_b<<<cdiv(tot, 256), 256, 0, stream>>>(Waa[l] + (size_t)cin * cout,
                                                        packW[2 * l + 1], cin, cout);
    }

    // ---- equivariant layers ----
    for (int l = 0; l < 3; ++l) {
        const int cin = ci_[l], cout = co_[l];
        const long cc = (long)cin * cout;
        const _Float16* Ain = (l == 0) ? nullptr : Abuf[(l == 1) ? 0 : 1];
        const float*    Din = (l == 0) ? x_diag : Dbuf[(l == 1) ? 0 : 1];
        _Float16* Aout = Abuf[(l == 1) ? 1 : 0];
        float*    Dout = Dbuf[(l == 1) ? 1 : 0];

        fill0<<<cdiv(zn, 256), 256, 0, stream>>>(Z, zn);

        if (l == 0)
            pool_edges_f32c7<<<cdiv(E, 256), 256, 0, stream>>>(x_adj, erow, ecol, rs, cs, ta, E);
        else if (l == 1)
            pool_edges_f16<32><<<cdiv(E, 256), 256, 0, stream>>>(Ain, erow, ecol, rs, cs, ta, E);
        else
            pool_edges_f16<64><<<cdiv(E, 256), 256, 0, stream>>>(Ain, erow, ecol, rs, cs, ta, E);
        pool_nodes<<<cdiv(N, 256), 256, 0, stream>>>(Din, td, N, cin);

        cvec_kernel<<<1, 64, 0, stream>>>(ta, td, Waa[l] + 6 * cc, Wda[l] + 2 * cc, ba[l], cvA, cin, cout);
        cvec_kernel<<<1, 64, 0, stream>>>(ta, td, Wad[l] + 2 * cc, Wdd[l] + 1 * cc, bd[l], cvD, cin, cout);

        node_terms_kernel<<<cdiv((long)N * cout, 256), 256, 0, stream>>>(
            rs, cs, Din,
            Waa[l] + 2 * cc, Waa[l] + 4 * cc, Wda[l],
            Waa[l] + 3 * cc, Waa[l] + 5 * cc, Wda[l] + 1 * cc,
            Wad[l],          Wad[l] + 1 * cc, Wdd[l],
            cvD, Rrow, Rcol, Dout, N, cin, cout);

        relu_stats_nodes<<<cdiv((long)N * cout, 256), 256, 0, stream>>>(
            Dout, dstat, (long)N * cout, cout);

        if (l == 0)
            edge_wmma_kernel<7, 32, false><<<E / 64, 128, 0, stream>>>(
                x_adj, packW[0], packW[1], Rrow, Rcol, cvA, erow, ecol, tidx, Aout, astat);
        else if (l == 1)
            edge_wmma_kernel<32, 64, true><<<E / 64, 128, 0, stream>>>(
                Ain, packW[2], packW[3], Rrow, Rcol, cvA, erow, ecol, tidx, Aout, astat);
        else
            edge_wmma_kernel<64, 32, true><<<E / 64, 128, 0, stream>>>(
                Ain, packW[4], packW[5], Rrow, Rcol, cvA, erow, ecol, tidx, Aout, astat);

        finalize_stats<<<1, 64, 0, stream>>>(astat, dstat, statfin, (float)E, (float)N, cout);
        norm_edges<<<cdiv((long)E * cout / 8, 256), 256, 0, stream>>>(
            Aout, statfin, (long)E * cout / 8, cout);
        norm_nodes<<<cdiv((long)N * cout, 256), 256, 0, stream>>>(
            Dout, statfin, (long)N * cout, cout);
    }

    // ---- final node head ----
    fill0<<<cdiv(zn, 256), 256, 0, stream>>>(Z, zn);
    pool_edges_f16<32><<<cdiv(E, 256), 256, 0, stream>>>(Abuf[0], erow, ecol, rs, cs, ta, E);
    pool_nodes<<<cdiv(N, 256), 256, 0, stream>>>(Dbuf[0], td, N, 32);
    cvec_kernel<<<1, 64, 0, stream>>>(ta, td, W4ad + 2 * 32 * 7, W4dd + 1 * 32 * 7, b4d, cvD, 32, 7);
    logits_kernel<<<cdiv(N, 128), 128, 0, stream>>>(
        rs, cs, Dbuf[0], W4ad, W4ad + 32 * 7, W4dd, cvD, (float*)d_out, N);
}
